// SequenceModel_31550829756933
// MI455X (gfx1250) — matrix operand order repaired and checked by reading, not compiled, not run on hardware
//
#include <hip/hip_runtime.h>
#include <math.h>

#define Bsz 64
#define Tn  4096
#define Fn  128
#define Hn  512
#define On  10
#define DTc 0.1f
#define NWG 32          // H/16 column tiles, one workgroup each
#define WG_THREADS 128  // 4 waves (wave32) = 4 row tiles of B=64

typedef __attribute__((ext_vector_type(16))) __bf16 v16bf;
typedef __attribute__((ext_vector_type(8)))  __bf16 v8bf;
typedef __attribute__((ext_vector_type(8)))  float  v8f;
typedef __attribute__((ext_vector_type(4)))  float  v4f;

// explicit global (address_space(1)) pointer types: force global_load/global_store
typedef float  __attribute__((address_space(1)))* as1_fp;
typedef const float  __attribute__((address_space(1)))* as1_cfp;
typedef __bf16 __attribute__((address_space(1)))* as1_bp;
typedef const v4f  __attribute__((address_space(1)))* as1_cv4fp;
typedef const v8bf __attribute__((address_space(1)))* as1_cv8bfp;

__device__ __forceinline__ v4f   ldg_v4f (const float* p)  { return *(as1_cv4fp)p; }
__device__ __forceinline__ v8bf  ldg_v8bf(const __bf16* p) { return *(as1_cv8bfp)p; }
__device__ __forceinline__ float ldg_f   (const float* p)  { return *(as1_cfp)p; }
__device__ __forceinline__ void  stg_f   (float* p, float v)   { *(as1_fp)p = v; }
__device__ __forceinline__ void  stg_bf  (__bf16* p, __bf16 v) { *(as1_bp)p = v; }

// workspace layout (bytes)
#define WS_SB  0                       // state bf16, two buffers (2 x 64x512)
#define WS_SF  (2*Bsz*Hn*2)            // final state f32 (64x512), for readout
#define WS_BAR (2*Bsz*Hn*2 + Bsz*Hn*4) // [count, generation]

__global__ void ctrnn_init(const float* __restrict__ ic, char* __restrict__ ws) {
    int idx = blockIdx.x * blockDim.x + threadIdx.x;
    __bf16* Sb0 = (__bf16*)(ws + WS_SB);
    unsigned* bar = (unsigned*)(ws + WS_BAR);
    if (idx < Bsz * Hn) {
        // broadcast trainable IC over batch (bf16 copy feeds the first GEMM)
        stg_bf(Sb0 + idx, (__bf16)ldg_f(ic + (idx & (Hn - 1))));
    }
    if (idx < 2) bar[idx] = 0;             // reset global barrier every launch
}

// fast, saturating tanh: copysign((1-e)/(1+e), x), e = exp(-2|x|) in (0,1]
__device__ __forceinline__ float fast_tanh(float x) {
    float e = __expf(-2.0f * fabsf(x));
    return copysignf((1.0f - e) / (1.0f + e), x);
}

// load + bf16-pack the 4 A-fragments of one x_t row slice (lane view)
__device__ __forceinline__ void load_xfrags(const float* xr, v16bf xf[4]) {
    #pragma unroll
    for (int k = 0; k < 4; ++k) {
        v4f f0 = ldg_v4f(xr + 32 * k);
        v4f f1 = ldg_v4f(xr + 32 * k + 4);
        v4f f2 = ldg_v4f(xr + 32 * k + 16);
        v4f f3 = ldg_v4f(xr + 32 * k + 20);
        v16bf a;
        #pragma unroll
        for (int e = 0; e < 4; ++e) {
            a[e]      = (__bf16)f0[e];
            a[e + 4]  = (__bf16)f1[e];
            a[e + 8]  = (__bf16)f2[e];
            a[e + 12] = (__bf16)f3[e];
        }
        xf[k] = a;
    }
}

__global__ __launch_bounds__(WG_THREADS)
void ctrnn_scan(const float* __restrict__ x,    // (B,T,F)
                const float* __restrict__ ic,   // (H)
                const float* __restrict__ Win,  // (F,H)
                const float* __restrict__ Wrec, // (H,H)
                const float* __restrict__ bvec, // (H)
                const float* __restrict__ tau,  // (H)
                char* __restrict__ ws)
{
    __shared__ __attribute__((aligned(32))) __bf16 ldsBrec[16 * 32 * 16]; // 16KB: W_rec slice, fragment-major
    __shared__ __attribute__((aligned(32))) __bf16 ldsBin [ 4 * 32 * 16]; //  4KB: W_in slice (staging)

    const int tid   = threadIdx.x;
    const int lane  = tid & 31;
    const int wave  = tid >> 5;             // 0..3 -> M (batch) tile
    const int n0    = blockIdx.x * 16;      // N (neuron) tile base
    const int khalf = lane >> 4;            // 0: lanes 0-15, 1: lanes 16-31
    const int col   = n0 + (lane & 15);     // C/D column of this lane
    const int row   = wave * 16 + (lane & 15); // A-matrix row (batch index)

    // Pre-swizzle B matrices into LDS fragment-major order.
    // B layout (32x16 bf16): lane = N + 16*Khalf; element e = K 8*Khalf + (e&7) + 16*(e>>3).
    for (int idx = tid; idx < 16 * 32 * 16; idx += WG_THREADS) {
        int e = idx & 15, L = (idx >> 4) & 31, k = idx >> 9;
        int krow = 32 * k + 8 * (L >> 4) + (e & 7) + 16 * (e >> 3);
        ldsBrec[idx] = (__bf16)ldg_f(Wrec + krow * Hn + n0 + (L & 15));
    }
    for (int idx = tid; idx < 4 * 32 * 16; idx += WG_THREADS) {
        int e = idx & 15, L = (idx >> 4) & 31, k = idx >> 9;
        int krow = 32 * k + 8 * (L >> 4) + (e & 7) + 16 * (e >> 3);
        ldsBin[idx] = (__bf16)ldg_f(Win + krow * Hn + n0 + (L & 15));
    }
    __syncthreads();

    // W_in B-fragments are step-invariant: keep them in registers for the scan.
    v16bf binreg[4];
    #pragma unroll
    for (int k = 0; k < 4; ++k)
        binreg[k] = *(const v16bf*)&ldsBin[(k * 32 + lane) * 16];

    const float bcol = ldg_f(bvec + col);
    const float acol = DTc / ldg_f(tau + col);  // per-neuron leak rate

    unsigned* bar = (unsigned*)(ws + WS_BAR);
    __bf16* Sb = (__bf16*)(ws + WS_SB);   // [cur][64][512] bf16 (cross-WG)
    float*  SfFin = (float*)(ws + WS_SF); // final f32 state for readout

    // Persistent register state: this lane owns C/D elements (rbase+r, col).
    const int rbase = wave * 16 + 8 * khalf;
    float sreg[8];
    {
        float s0 = ldg_f(ic + col);
        #pragma unroll
        for (int r = 0; r < 8; ++r) sreg[r] = s0;
    }

    // Software-pipelined x fragments (t = 0 preloaded).
    const float* xbase = x + ((size_t)row * Tn) * Fn + 8 * khalf;
    v16bf xf[4];
    load_xfrags(xbase, xf);

    int cur = 0;
    for (int t = 0; t < Tn; ++t) {
        // ---- input projection x_t @ W_in: 4 register-only WMMAs ----
        v8f acc = {};
        #pragma unroll
        for (int k = 0; k < 4; ++k)
            acc = __builtin_amdgcn_wmma_f32_16x16x32_bf16(false, xf[k], false, binreg[k],
                                                          (short)0, acc, false, false);

        // ---- recurrence: S @ W_rec (K = 512, 16 chunks, fully unrolled) ----
        const __bf16* srow = Sb + cur * (Bsz * Hn) + row * Hn + 8 * khalf;
        #pragma unroll
        for (int k = 0; k < 16; ++k) {
            v8bf a0 = ldg_v8bf(srow + 32 * k);        // K 0..7   (+8 if khalf)
            v8bf a1 = ldg_v8bf(srow + 32 * k + 16);   // K 16..23 (+8 if khalf)
            v16bf afrag;
            #pragma unroll
            for (int e = 0; e < 8; ++e) { afrag[e] = a0[e]; afrag[e + 8] = a1[e]; }
            v16bf bfrag = *(const v16bf*)&ldsBrec[(k * 32 + lane) * 16];
            acc = __builtin_amdgcn_wmma_f32_16x16x32_bf16(false, afrag, false, bfrag,
                                                          (short)0, acc, false, false);
        }

        // ---- prefetch/pipeline x for t+1 (independent of state: hides behind
        //      the WMMA chain and the barrier spin) ----
        if (t + 1 < Tn) load_xfrags(xbase + (size_t)(t + 1) * Fn, xf);
        if (t + 2 < Tn) __builtin_prefetch(xbase + (size_t)(t + 2) * Fn, 0, 1);

        // ---- CTRNN update in registers; publish bf16 state for other WGs ----
        __bf16* Sbn = Sb + (cur ^ 1) * (Bsz * Hn) + rbase * Hn + col;
        #pragma unroll
        for (int r = 0; r < 8; ++r) {
            float sn = sreg[r] + acol * (fast_tanh(acc[r] + bcol) - sreg[r]);
            sreg[r] = sn;
            stg_bf(Sbn + r * Hn, (__bf16)sn);
        }
        if (t == Tn - 1) {                     // f32 state only needed at the end
            #pragma unroll
            for (int r = 0; r < 8; ++r) stg_f(SfFin + (rbase + r) * Hn + col, sreg[r]);
        }

        // ---- grid-wide barrier: release, arrive, spin on generation, acquire ----
        __threadfence();
        if (tid == 0) {
            if (atomicAdd(bar, 1u) == NWG - 1) {
                *(volatile unsigned*)bar = 0;
                __threadfence();
                atomicAdd(bar + 1, 1u);
            }
            while (*(volatile unsigned*)(bar + 1) < (unsigned)(t + 1))
                __builtin_amdgcn_s_sleep(1);
        }
        __syncthreads();
        __threadfence();
        cur ^= 1;
    }
}

__global__ void ctrnn_readout(const char* __restrict__ ws,
                              const float* __restrict__ rw,   // (H,O)
                              const float* __restrict__ rb,   // (O)
                              float* __restrict__ out)        // (B,O)
{
    const float* S = (const float*)(ws + WS_SF);   // final f32 state
    int i = blockIdx.x;      // batch row
    int o = threadIdx.x;     // output index
    if (o < On) {
        float acc = rb[o];
        for (int h = 0; h < Hn; ++h) acc += ldg_f(S + i * Hn + h) * ldg_f(rw + h * On + o);
        out[i * On + o] = acc;
    }
}

extern "C" void kernel_launch(void* const* d_in, const int* in_sizes, int n_in,
                              void* d_out, int out_size, void* d_ws, size_t ws_size,
                              hipStream_t stream) {
    (void)in_sizes; (void)n_in; (void)out_size; (void)ws_size;
    const float* x    = (const float*)d_in[0];
    const float* ic   = (const float*)d_in[1];
    const float* Win  = (const float*)d_in[2];
    const float* Wrec = (const float*)d_in[3];
    const float* b    = (const float*)d_in[4];
    const float* tau  = (const float*)d_in[5];
    const float* rw   = (const float*)d_in[6];
    const float* rb   = (const float*)d_in[7];
    float* out = (float*)d_out;
    char*  ws  = (char*)d_ws;

    ctrnn_init<<<(Bsz * Hn + 255) / 256, 256, 0, stream>>>(ic, ws);
    ctrnn_scan<<<NWG, WG_THREADS, 0, stream>>>(x, ic, Win, Wrec, b, tau, ws);
    ctrnn_readout<<<Bsz, 32, 0, stream>>>(ws, rw, rb, out);
}
